// GNNModel_48155173323172
// MI455X (gfx1250) — matrix operand order repaired and checked by reading, not compiled, not run on hardware
//
#include <hip/hip_runtime.h>

#define N_NODES   100000
#define N_EDGES   1600000
#define IN_CH     128
#define HIDDEN    64
#define N_CLASSES 40

typedef __attribute__((ext_vector_type(16))) __bf16 v16bf;
typedef __attribute__((ext_vector_type(8)))  float  v8f;

// round-to-nearest-even f32 -> bf16
__device__ __forceinline__ __bf16 cvt_bf16(float f) {
    unsigned u = __builtin_bit_cast(unsigned, f);
    u += 0x7FFFu + ((u >> 16) & 1u);
    unsigned short h = (unsigned short)(u >> 16);
    return __builtin_bit_cast(__bf16, h);
}

// ---------------------------------------------------------------------------
// deg[i] = 1 (self loop), +1 per edge dst, then dinv = rsqrt(deg)
// ---------------------------------------------------------------------------
__global__ void set_one_kernel(float* __restrict__ deg, int n) {
    int i = blockIdx.x * blockDim.x + threadIdx.x;
    if (i < n) deg[i] = 1.0f;
}

__global__ void degree_kernel(const long long* __restrict__ dst,
                              float* __restrict__ deg) {
    int e = blockIdx.x * blockDim.x + threadIdx.x;   // grid sized exactly E
    int d = (int)dst[e];
    unsafeAtomicAdd(&deg[d], 1.0f);                  // global_atomic_add_f32
}

__global__ void rsqrt_kernel(float* __restrict__ dinv, int n) {
    int i = blockIdx.x * blockDim.x + threadIdx.x;
    if (i < n) dinv[i] = rsqrtf(dinv[i]);            // deg >= 1 always
}

// ---------------------------------------------------------------------------
// GEMM  C[M x N] = A[M x K] * W[K x N]   (row-major, bf16 WMMA, f32 accum)
//  - W staged once per block into LDS, bf16, transposed [n][k], zero-padded
//    to NPAD = NT*16 columns -> B fragments are unguarded ds_load_b128 pairs.
//  - One wave per 16-row strip of A; A fragment reused across NT column
//    tiles -> NT v_wmma per K-step, K-loop fully unrolled (K template).
//  Fragment layouts (wave32): A/B use contiguous k order, k = 16*half + s.
// ---------------------------------------------------------------------------
template<int NT, int K>
__global__ void gemm_bf16_wmma(const float* __restrict__ A,
                               const float* __restrict__ W,
                               float* __restrict__ C,
                               int N, int mTiles) {
    constexpr int NPAD = NT * 16;
    __shared__ __bf16 sWt[NPAD * K];                 // [n][k], max 16 KB

    for (int idx = threadIdx.x; idx < NPAD * K; idx += blockDim.x) {
        int n = idx / K;                             // constant divisor
        int k = idx - n * K;
        float w = (n < N) ? W[(size_t)k * N + n] : 0.0f;
        sWt[idx] = cvt_bf16(w);
    }
    __syncthreads();

    const int wave = blockIdx.x * (blockDim.x >> 5) + (threadIdx.x >> 5);
    if (wave >= mTiles) return;                      // wave-uniform
    const int lane = threadIdx.x & 31;
    const int half = lane >> 4;                      // 0: lanes 0-15, 1: 16-31
    const int l15  = lane & 15;
    const int rowA = wave * 16 + l15;

    v8f acc[NT];
#pragma unroll
    for (int t = 0; t < NT; ++t) acc[t] = (v8f){};

#pragma unroll
    for (int k0 = 0; k0 < K; k0 += 32) {
        // A 16x32 bf16: lanes<16 hold K k0+0..15,
        //               lanes>=16 hold k0+16..31.
        v16bf a;
        const float* ap = A + (size_t)rowA * K + k0 + half * 16;
#pragma unroll
        for (int j = 0; j < 8; ++j) {
            a[j]     = cvt_bf16(ap[j]);
            a[8 + j] = cvt_bf16(ap[8 + j]);
        }
        const int kB = k0 + half * 16;               // B: lanes<16 K lo, >=16 K hi
#pragma unroll
        for (int t = 0; t < NT; ++t) {
            v16bf b = *(const v16bf*)&sWt[(t * 16 + l15) * K + kB]; // 2x ds_load_b128
            acc[t] = __builtin_amdgcn_wmma_f32_16x16x32_bf16(
                         false, a, false, b, (short)0, acc[t], false, false);
        }
    }

#pragma unroll
    for (int t = 0; t < NT; ++t) {
        int col = t * 16 + l15;
        if (col < N) {                               // only trims GEMM2 last tile
            float* cp = C + (size_t)(wave * 16 + half * 8) * N + col;
#pragma unroll
            for (int r = 0; r < 8; ++r) cp[(size_t)r * N] = acc[t][r];
        }
    }
}

// ---------------------------------------------------------------------------
// Edge scatter layer 1: one thread per (edge, channel<64).  Gathers from hw
// and atomics into hagg are coalesced over channels; both buffers fit in the
// 192 MB L2, so atomics execute at the L2 atomic units.
// ---------------------------------------------------------------------------
__global__ void scatter64_kernel(const long long* __restrict__ src,
                                 const long long* __restrict__ dst,
                                 const float* __restrict__ dinv,
                                 const float* __restrict__ hw,
                                 float* __restrict__ hagg) {
    unsigned idx = blockIdx.x * blockDim.x + threadIdx.x;   // grid exactly E*64
    int e = (int)(idx >> 6);
    int c = (int)(idx & 63u);
    int s = (int)src[e];
    int d = (int)dst[e];
    float w = dinv[s] * dinv[d];
    unsafeAtomicAdd(&hagg[(size_t)d * HIDDEN + c], hw[(size_t)s * HIDDEN + c] * w);
}

// self-loop contribution + bias + ReLU (self loops need no atomics)
__global__ void bias_relu_kernel(float* __restrict__ hagg,
                                 const float* __restrict__ hw,
                                 const float* __restrict__ dinv,
                                 const float* __restrict__ b1) {
    unsigned idx = blockIdx.x * blockDim.x + threadIdx.x;   // grid exactly N*64
    int i = (int)(idx >> 6);
    int c = (int)(idx & 63u);
    float di = dinv[i];
    float v = hagg[idx] + hw[idx] * di * di + b1[c];
    hagg[idx] = fmaxf(v, 0.0f);
}

// Edge scatter layer 2: one thread per (edge, channel<40) -> d_out
__global__ void scatter40_kernel(const long long* __restrict__ src,
                                 const long long* __restrict__ dst,
                                 const float* __restrict__ dinv,
                                 const float* __restrict__ hw,
                                 float* __restrict__ out) {
    unsigned idx = blockIdx.x * blockDim.x + threadIdx.x;   // grid exactly E*40
    int e = (int)(idx / 40u);
    int c = (int)(idx - 40u * (unsigned)e);
    int s = (int)src[e];
    int d = (int)dst[e];
    float w = dinv[s] * dinv[d];
    unsafeAtomicAdd(&out[(size_t)d * N_CLASSES + c], hw[(size_t)s * N_CLASSES + c] * w);
}

__global__ void bias2_kernel(float* __restrict__ out,
                             const float* __restrict__ hw,
                             const float* __restrict__ dinv,
                             const float* __restrict__ b2) {
    unsigned idx = blockIdx.x * blockDim.x + threadIdx.x;   // grid exactly N*40
    int i = (int)(idx / 40u);
    int c = (int)(idx - 40u * (unsigned)i);
    float di = dinv[i];
    out[idx] += hw[idx] * di * di + b2[c];
}

// ---------------------------------------------------------------------------
extern "C" void kernel_launch(void* const* d_in, const int* in_sizes, int n_in,
                              void* d_out, int out_size, void* d_ws, size_t ws_size,
                              hipStream_t stream) {
    const float*     x  = (const float*)d_in[0];
    const long long* ei = (const long long*)d_in[1];   // int64 edge_index [2, E]
    const float*     W1 = (const float*)d_in[2];
    const float*     b1 = (const float*)d_in[3];
    const float*     W2 = (const float*)d_in[4];
    const float*     b2 = (const float*)d_in[5];
    float* out = (float*)d_out;
    float* ws  = (float*)d_ws;

    // workspace layout (floats): dinv | hw1 | hagg | hw2   (~67.6 MB total)
    float* dinv = ws;                                  // 100000 (padded to 100096)
    float* hw1  = ws + 100096;                         // 100000*64
    float* hagg = hw1 + (size_t)N_NODES * HIDDEN;      // 100000*64
    float* hw2  = hagg + (size_t)N_NODES * HIDDEN;     // 100000*40

    const long long* src = ei;                         // row 0
    const long long* dst = ei + N_EDGES;               // row 1

    // zero the accumulation targets (graph-capturable memset nodes)
    hipMemsetAsync(hagg, 0, (size_t)N_NODES * HIDDEN    * sizeof(float), stream);
    hipMemsetAsync(out,  0, (size_t)N_NODES * N_CLASSES * sizeof(float), stream);

    // degrees (self-loop via init to 1) -> dinv
    set_one_kernel<<<(N_NODES + 255) / 256, 256, 0, stream>>>(dinv, N_NODES);
    degree_kernel<<<N_EDGES / 256, 256, 0, stream>>>(dst, dinv);
    rsqrt_kernel<<<(N_NODES + 255) / 256, 256, 0, stream>>>(dinv, N_NODES);

    const int mTiles = N_NODES / 16;                   // 6250, exact
    const int gemmBlocks = (mTiles + 7) / 8;           // 8 waves / 256-thread block

    // layer 1: hw1 = x @ W1   (NT=4 column tiles, K=128)
    gemm_bf16_wmma<4, IN_CH><<<gemmBlocks, 256, 0, stream>>>(x, W1, hw1, HIDDEN, mTiles);
    scatter64_kernel<<<(unsigned)N_EDGES * 64u / 256u, 256, 0, stream>>>(src, dst, dinv, hw1, hagg);
    bias_relu_kernel<<<(unsigned)N_NODES * 64u / 256u, 256, 0, stream>>>(hagg, hw1, dinv, b1);

    // layer 2: hw2 = hagg @ W2   (NT=3 column tiles, K=64, N=40 trimmed on store)
    gemm_bf16_wmma<3, HIDDEN><<<gemmBlocks, 256, 0, stream>>>(hagg, W2, hw2, N_CLASSES, mTiles);
    scatter40_kernel<<<(unsigned)N_EDGES * 40u / 256u, 256, 0, stream>>>(src, dst, dinv, hw2, out);
    bias2_kernel<<<(unsigned)N_NODES * 40u / 256u, 256, 0, stream>>>(out, hw2, dinv, b2);
}
